// BERTCRF_21672404975887
// MI455X (gfx1250) — matrix operand order repaired and checked by reading, not compiled, not run on hardware
//
#include <hip/hip_runtime.h>
#include <hip/hip_bf16.h>

// ---------------------------------------------------------------------------
// BiLSTM + CRF loss for MI455X (gfx1250, wave32, WMMA, async global->LDS).
//  K1: xg = X*W_ih^T + b   via bf16 WMMA (both directions)   [batch GEMM]
//  K2: per-timestep gate GEMM h*W_hh^T via bf16 WMMA + LSTM cell update
//  K3: emissions = [h_f|h_b]*W_cls^T + b_cls                 [VALU]
//  K4: CRF numerator + forward-algorithm logsumexp, one lane per batch
// ---------------------------------------------------------------------------

typedef __attribute__((ext_vector_type(16))) __bf16 v16bf;
typedef __attribute__((ext_vector_type(8)))  __bf16 v8bf;
typedef __attribute__((ext_vector_type(8)))  float  v8f;
typedef __attribute__((ext_vector_type(4)))  int    v4i;

#define BB 32
#define SS 512
#define DD 768
#define GG 3072           // 4*DD
#define TT 9
#define MS (BB*SS)        // 16384

// --- gfx1250 async global->LDS path (guarded; falls back to ld+ds_store) ---
#if defined(__AMDGCN__) && __has_builtin(__builtin_amdgcn_global_load_async_to_lds_b128)
#define HAVE_ASYNC_LDS 1
#else
#define HAVE_ASYNC_LDS 0
#endif

#if HAVE_ASYNC_LDS
typedef __attribute__((address_space(1))) v4i* gas_v4i_ptr;   // global (AS1)
typedef __attribute__((address_space(3))) v4i* lds_v4i_ptr;   // LDS (AS3)
#endif

__device__ __forceinline__ void async_copy16(void* lds_dst, const void* gsrc) {
#if HAVE_ASYNC_LDS
    __builtin_amdgcn_global_load_async_to_lds_b128(
        (gas_v4i_ptr)(gsrc), (lds_v4i_ptr)(lds_dst), 0, 0);
#else
    *(v8bf*)lds_dst = *(const v8bf*)gsrc;
#endif
}

__device__ __forceinline__ void wait_async0() {
#if HAVE_ASYNC_LDS
#if __has_builtin(__builtin_amdgcn_s_wait_asynccnt)
    __builtin_amdgcn_s_wait_asynccnt(0);
#else
    asm volatile("s_wait_asynccnt 0x0" ::: "memory");
#endif
#endif
}

__device__ __forceinline__ __bf16 f2bf(float f) {
    unsigned u = __builtin_bit_cast(unsigned, f);
    u += 0x7fffu + ((u >> 16) & 1u);            // round-to-nearest-even
    unsigned short h = (unsigned short)(u >> 16);
    return __builtin_bit_cast(__bf16, h);
}
__device__ __forceinline__ float sigmoidf(float x) { return 1.0f / (1.0f + __expf(-x)); }

// ---------------------------------------------------------------------------
__global__ void k_f32_to_bf16(const float* __restrict__ src, __bf16* __restrict__ dst, int n) {
    int i = blockIdx.x * blockDim.x + threadIdx.x;
    if (i < n) dst[i] = f2bf(src[i]);
}

__global__ void k_zero(unsigned int* __restrict__ p, int nwords) {
    int i = blockIdx.x * blockDim.x + threadIdx.x;
    if (i < nwords) p[i] = 0u;
}

// ---------------------------------------------------------------------------
// K1: C[M=16384][N=3072] = A[M][K=768](bf16) * W[N][K](bf16)^T + bias, f32 out
// block = 256 thr (8 waves), tile 128x128, K-step 32. Wave = 64x32 subtile
// (4 M-tiles x 2 N-tiles of 16x16), accumulators 8 x v8f.
__global__ __launch_bounds__(256)
void k_gemm_xg(const __bf16* __restrict__ A, const __bf16* __restrict__ W,
               const float* __restrict__ bias, float* __restrict__ C)
{
    __shared__ __align__(16) __bf16 As[128][40];   // +8 pad -> 16B-aligned rows
    __shared__ __align__(16) __bf16 Bs[128][40];

    const int m0 = blockIdx.x * 128;
    const int n0 = blockIdx.y * 128;
    const int tid  = threadIdx.x;
    const int lane = tid & 31;
    const int wave = tid >> 5;
    const int wm = wave & 1;        // M half (64 rows)
    const int wn = wave >> 1;       // N quarter (32 cols)

    const v8f vzero = {0.f,0.f,0.f,0.f,0.f,0.f,0.f,0.f};
    v8f acc[4][2];
#pragma unroll
    for (int i = 0; i < 4; ++i) { acc[i][0] = vzero; acc[i][1] = vzero; }

    const int lr = tid >> 1;            // 0..127
    const int lc = (tid & 1) * 16;      // 0 or 16

    for (int k0 = 0; k0 < DD; k0 += 32) {
        // stage 128x32 tiles of A and W: global -> LDS (async on gfx1250)
        async_copy16(&As[lr][lc],     &A[(size_t)(m0 + lr) * DD + k0 + lc]);
        async_copy16(&As[lr][lc + 8], &A[(size_t)(m0 + lr) * DD + k0 + lc + 8]);
        async_copy16(&Bs[lr][lc],     &W[(size_t)(n0 + lr) * DD + k0 + lc]);
        async_copy16(&Bs[lr][lc + 8], &W[(size_t)(n0 + lr) * DD + k0 + lc + 8]);
        if (k0 + 32 < DD) {  // warm L2 for next K tile (global_prefetch_b8)
            __builtin_prefetch(&A[(size_t)(m0 + lr) * DD + k0 + 32 + lc], 0, 1);
            __builtin_prefetch(&W[(size_t)(n0 + lr) * DD + k0 + 32 + lc], 0, 1);
        }
        wait_async0();
        __syncthreads();

        // B fragments: lane<16 -> K {0..7,16..23}, lane>=16 -> K {8..15,24..31}, col = lane%16
        v16bf bfr[2];
#pragma unroll
        for (int nt = 0; nt < 2; ++nt) {
            int n  = wn * 32 + nt * 16 + (lane & 15);
            int kk = (lane < 16) ? 0 : 8;
            *(v8bf*)&bfr[nt]        = *(const v8bf*)&Bs[n][kk];
            ((v8bf*)&bfr[nt])[1]    = *(const v8bf*)&Bs[n][kk + 16];
        }
        // A fragments + WMMA
#pragma unroll
        for (int mt = 0; mt < 4; ++mt) {
            int m  = wm * 64 + mt * 16 + (lane & 15);
            int kb = (lane < 16) ? 0 : 8;   // K {0..7,16..23} / {8..15,24..31}
            v16bf afr;
            *(v8bf*)&afr     = *(const v8bf*)&As[m][kb];
            ((v8bf*)&afr)[1] = *(const v8bf*)&As[m][kb + 16];
            acc[mt][0] = __builtin_amdgcn_wmma_f32_16x16x32_bf16(
                false, afr, false, bfr[0], (short)0, acc[mt][0], false, false);
            acc[mt][1] = __builtin_amdgcn_wmma_f32_16x16x32_bf16(
                false, afr, false, bfr[1], (short)0, acc[mt][1], false, false);
        }
        __syncthreads();
    }

    // epilogue: D layout lane<16 -> M 0..7, lane>=16 -> M 8..15, col = lane%16
#pragma unroll
    for (int nt = 0; nt < 2; ++nt) {
        int n = n0 + wn * 32 + nt * 16 + (lane & 15);
        float bv = bias[n];
#pragma unroll
        for (int mt = 0; mt < 4; ++mt) {
            int mb = m0 + wm * 64 + mt * 16 + ((lane < 16) ? 0 : 8);
#pragma unroll
            for (int r = 0; r < 8; ++r)
                C[(size_t)(mb + r) * GG + n] = acc[mt][nt][r] + bv;
        }
    }
}

// ---------------------------------------------------------------------------
// K2: one timestep, both directions (blockIdx.z). Each block owns 32 hidden
// units -> 128 gate columns (4 gates x 32). h_prev (32x768 bf16) staged in LDS
// once (async); W_hh rows streamed from L2 straight into B fragments.
// 8 waves: wave = (mt in 0..1, ntp in 0..3), 2 WMMA per K-iter, 24 K-iters.
// Gate pre-activations go through LDS, then 1024 (j,m) cell updates.
__global__ __launch_bounds__(256)
void k_lstm_step(const __bf16* __restrict__ whhF, const __bf16* __restrict__ whhB,
                 const float* __restrict__ xgF,  const float* __restrict__ xgB,
                 const __bf16* __restrict__ hinF, const __bf16* __restrict__ hinB,
                 __bf16* __restrict__ houtF, __bf16* __restrict__ houtB,
                 float* __restrict__ cF, float* __restrict__ cB,
                 float* __restrict__ hsF, float* __restrict__ hsB, int t)
{
    constexpr int HPAD = DD + 8;                      // 776 halves/row
    constexpr int SM_H = 32 * HPAD * 2;               // 49664 B
    constexpr int SM_G = 128 * 36 * 4;                // 18432 B
    constexpr int SM   = (SM_H > SM_G) ? SM_H : SM_G;
    __shared__ __align__(16) char smem[SM];
    auto Hs   = (__bf16(*)[HPAD])smem;                // phase 1
    auto gbuf = (float(*)[36])smem;                   // phase 2 (aliased)

    const int dir = blockIdx.z;
    const __bf16* whh = dir ? whhB : whhF;
    const float*  xg  = dir ? xgB  : xgF;
    const __bf16* hin = dir ? hinB : hinF;
    __bf16* hout = dir ? houtB : houtF;
    float*  cst  = dir ? cB : cF;
    float*  hs   = dir ? hsB : hsF;
    const int s  = dir ? (SS - 1 - t) : t;            // backward consumes x[::-1]
    const int j0 = blockIdx.x * 32;

    const int tid  = threadIdx.x;
    const int lane = tid & 31;
    const int wave = tid >> 5;

    // stage h_prev: 32 x 768 bf16 (async global -> LDS)
    {
        int r  = tid >> 3;
        int cb = (tid & 7) * 96;
#pragma unroll
        for (int q = 0; q < 12; ++q)
            async_copy16(&Hs[r][cb + q * 8], &hin[r * DD + cb + q * 8]);
    }
    wait_async0();
    __syncthreads();

    const int mt  = wave & 1;                          // batch half
    const int ntp = wave >> 1;                         // pair of N-tiles
    const int nl0 = (ntp * 2 + 0) * 16 + (lane & 15);  // local gate column
    const int nl1 = (ntp * 2 + 1) * 16 + (lane & 15);
    const __bf16* brow0 = whh + (size_t)((nl0 >> 5) * DD + j0 + (nl0 & 31)) * DD;
    const __bf16* brow1 = whh + (size_t)((nl1 >> 5) * DD + j0 + (nl1 & 31)) * DD;
    const int ksel = (lane < 16) ? 0 : 8;
    const int asel = (lane < 16) ? 0 : 8;
    const int mrow = mt * 16 + (lane & 15);

    const v8f vzero = {0.f,0.f,0.f,0.f,0.f,0.f,0.f,0.f};
    v8f acc0 = vzero, acc1 = vzero;

#pragma unroll 4
    for (int k0 = 0; k0 < DD; k0 += 32) {
        v16bf a, b0, b1;
        *(v8bf*)&a      = *(const v8bf*)&Hs[mrow][k0 + asel];
        ((v8bf*)&a)[1]  = *(const v8bf*)&Hs[mrow][k0 + asel + 16];
        *(v8bf*)&b0     = *(const v8bf*)&brow0[k0 + ksel];
        ((v8bf*)&b0)[1] = *(const v8bf*)&brow0[k0 + ksel + 16];
        *(v8bf*)&b1     = *(const v8bf*)&brow1[k0 + ksel];
        ((v8bf*)&b1)[1] = *(const v8bf*)&brow1[k0 + ksel + 16];
        acc0 = __builtin_amdgcn_wmma_f32_16x16x32_bf16(false, a, false, b0, (short)0, acc0, false, false);
        acc1 = __builtin_amdgcn_wmma_f32_16x16x32_bf16(false, a, false, b1, (short)0, acc1, false, false);
    }
    __syncthreads();            // Hs dead; re-use LDS for gate buffer

    {
        int mb = mt * 16 + ((lane < 16) ? 0 : 8);
#pragma unroll
        for (int r = 0; r < 8; ++r) {
            gbuf[nl0][mb + r] = acc0[r];
            gbuf[nl1][mb + r] = acc1[r];
        }
    }
    __syncthreads();

    // elementwise LSTM cell: 32 hidden x 32 batch = 1024 updates
#pragma unroll
    for (int q = 0; q < 4; ++q) {
        int idx = q * 256 + tid;
        int jj  = idx & 31;
        int m   = idx >> 5;
        size_t xb = (size_t)(m * SS + s) * GG + j0 + jj;
        float gi = gbuf[ 0 + jj][m] + xg[xb];
        float gf = gbuf[32 + jj][m] + xg[xb + DD];
        float gg = gbuf[64 + jj][m] + xg[xb + 2 * DD];
        float go = gbuf[96 + jj][m] + xg[xb + 3 * DD];
        int   ci = m * DD + j0 + jj;
        float c  = sigmoidf(gf) * cst[ci] + sigmoidf(gi) * tanhf(gg);
        float h  = sigmoidf(go) * tanhf(c);
        cst[ci] = c;
        hs[(size_t)(m * SS + s) * DD + j0 + jj] = h;
        hout[ci] = f2bf(h);
    }
}

// ---------------------------------------------------------------------------
// K3: emissions[row][t] = b_cls[t] + h_f[row].w[t,:768] + h_b[row].w[t,768:]
__global__ void k_emissions(const float* __restrict__ hsF, const float* __restrict__ hsB,
                            const float* __restrict__ wcls, const float* __restrict__ bcls,
                            float* __restrict__ em)
{
    int idx = blockIdx.x * blockDim.x + threadIdx.x;
    if (idx >= MS * TT) return;
    int row = idx / TT, tt = idx % TT;
    const float* wf = wcls + (size_t)tt * (2 * DD);
    const float* wb = wf + DD;
    const float* hf = hsF + (size_t)row * DD;
    const float* hb = hsB + (size_t)row * DD;
    float acc = bcls[tt];
#pragma unroll 8
    for (int d = 0; d < DD; ++d) acc += hf[d] * wf[d] + hb[d] * wb[d];
    em[idx] = acc;
}

// ---------------------------------------------------------------------------
// K4: CRF loss. One wave32, lane b handles batch b. 9-state forward algorithm.
__global__ __launch_bounds__(32)
void k_crf(const float* __restrict__ em, const int* __restrict__ tags,
           const unsigned char* __restrict__ mask,
           const float* __restrict__ trans, const float* __restrict__ startt,
           const float* __restrict__ endt, float* __restrict__ out)
{
    const int b = threadIdx.x;           // 0..31
    float tr[TT][TT];
#pragma unroll
    for (int i = 0; i < TT; ++i)
#pragma unroll
        for (int j = 0; j < TT; ++j) tr[i][j] = trans[i * TT + j];

    const size_t rb = (size_t)b * SS;
    float score[TT];
    int tg0 = tags[rb];
    float num = startt[tg0] + em[rb * TT + tg0];
#pragma unroll
    for (int j = 0; j < TT; ++j) score[j] = startt[j] + em[rb * TT + j];
    int cnt = mask[rb] ? 1 : 0;
    int tprev = tg0;

    for (int s = 1; s < SS; ++s) {
        const float* ems = em + (rb + s) * TT;
        int tc = tags[rb + s];
        bool mm = mask[rb + s];
        float mf = mm ? 1.f : 0.f;
        num += (tr[tprev][tc] + ems[tc]) * mf;
        tprev = tc;
        cnt += mm ? 1 : 0;
        float nxt[TT];
#pragma unroll
        for (int j = 0; j < TT; ++j) {
            float mx = -3.4e38f;
#pragma unroll
            for (int i = 0; i < TT; ++i) mx = fmaxf(mx, score[i] + tr[i][j]);
            float ssum = 0.f;
#pragma unroll
            for (int i = 0; i < TT; ++i) ssum += __expf(score[i] + tr[i][j] - mx);
            nxt[j] = mx + __logf(ssum) + ems[j];
        }
#pragma unroll
        for (int j = 0; j < TT; ++j) score[j] = mm ? nxt[j] : score[j];
    }
    int li = cnt - 1; if (li < 0) li = 0;
    num += endt[tags[rb + li]];

    float mx = -3.4e38f;
#pragma unroll
    for (int j = 0; j < TT; ++j) mx = fmaxf(mx, score[j] + endt[j]);
    float ssum = 0.f;
#pragma unroll
    for (int j = 0; j < TT; ++j) ssum += __expf(score[j] + endt[j] - mx);
    float den = mx + __logf(ssum);

    float llh  = num - den;
    float mcnt = (float)cnt;
    for (int o = 16; o > 0; o >>= 1) {
        llh  += __shfl_down(llh,  o, 32);
        mcnt += __shfl_down(mcnt, o, 32);
    }
    if (b == 0) out[0] = -(llh / mcnt);
}

// ---------------------------------------------------------------------------
extern "C" void kernel_launch(void* const* d_in, const int* in_sizes, int n_in,
                              void* d_out, int out_size, void* d_ws, size_t ws_size,
                              hipStream_t stream)
{
    (void)in_sizes; (void)n_in; (void)out_size; (void)ws_size;

    const float* x    = (const float*)d_in[0];
    const int*   tags = (const int*)d_in[1];
    const unsigned char* mask = (const unsigned char*)d_in[2];
    const float* wihF = (const float*)d_in[3];
    const float* whhF = (const float*)d_in[4];
    const float* bF   = (const float*)d_in[5];
    const float* wihB = (const float*)d_in[6];
    const float* whhB = (const float*)d_in[7];
    const float* bB   = (const float*)d_in[8];
    const float* wcls = (const float*)d_in[9];
    const float* bcls = (const float*)d_in[10];
    const float* trans= (const float*)d_in[11];
    const float* st   = (const float*)d_in[12];
    const float* et   = (const float*)d_in[13];

    char* ws = (char*)d_ws;
    size_t off = 0;
    auto carve = [&](size_t bytes) -> char* {
        char* p = ws + off;
        off += (bytes + 255) & ~(size_t)255;
        return p;
    };
    __bf16* x_bf    = (__bf16*)carve((size_t)MS * DD * 2);
    __bf16* wihF_bf = (__bf16*)carve((size_t)GG * DD * 2);
    __bf16* wihB_bf = (__bf16*)carve((size_t)GG * DD * 2);
    __bf16* whhF_bf = (__bf16*)carve((size_t)GG * DD * 2);
    __bf16* whhB_bf = (__bf16*)carve((size_t)GG * DD * 2);
    float*  xgF     = (float*)carve((size_t)MS * GG * 4);
    float*  xgB     = (float*)carve((size_t)MS * GG * 4);
    float*  hsF     = (float*)carve((size_t)MS * DD * 4);
    float*  hsB     = (float*)carve((size_t)MS * DD * 4);
    __bf16* hbf     = (__bf16*)carve((size_t)4 * BB * DD * 2);  // [dir][pingpong]
    float*  cstate  = (float*)carve((size_t)2 * BB * DD * 4);
    float*  emis    = (float*)carve((size_t)MS * TT * 4);

    // --- precision conversion (fp32 -> bf16 operands for WMMA) ---
    const int nX = MS * DD, nW = GG * DD;
    k_f32_to_bf16<<<(nX + 255) / 256, 256, 0, stream>>>(x,    x_bf,    nX);
    k_f32_to_bf16<<<(nW + 255) / 256, 256, 0, stream>>>(wihF, wihF_bf, nW);
    k_f32_to_bf16<<<(nW + 255) / 256, 256, 0, stream>>>(wihB, wihB_bf, nW);
    k_f32_to_bf16<<<(nW + 255) / 256, 256, 0, stream>>>(whhF, whhF_bf, nW);
    k_f32_to_bf16<<<(nW + 255) / 256, 256, 0, stream>>>(whhB, whhB_bf, nW);

    // --- zero recurrent state (h0, c0) every call ---
    k_zero<<<(4 * BB * DD / 2 + 255) / 256, 256, 0, stream>>>((unsigned*)hbf,    4 * BB * DD / 2);
    k_zero<<<(2 * BB * DD     + 255) / 256, 256, 0, stream>>>((unsigned*)cstate, 2 * BB * DD);

    // --- hoisted input GEMMs: xg = X*W_ih^T + b (both directions) ---
    dim3 gGemm(MS / 128, GG / 128);   // 128 x 24
    k_gemm_xg<<<gGemm, 256, 0, stream>>>(x_bf, wihF_bf, bF, xgF);
    k_gemm_xg<<<gGemm, 256, 0, stream>>>(x_bf, wihB_bf, bB, xgB);

    // --- 512 sequential LSTM steps, fwd+bwd concurrently (grid z = dir) ---
    __bf16* hF0 = hbf;
    __bf16* hF1 = hbf + BB * DD;
    __bf16* hB0 = hbf + 2 * BB * DD;
    __bf16* hB1 = hbf + 3 * BB * DD;
    float*  cF  = cstate;
    float*  cB  = cstate + BB * DD;
    for (int t = 0; t < SS; ++t) {
        const __bf16* hinF = (t & 1) ? hF1 : hF0;
        __bf16*      houtF = (t & 1) ? hF0 : hF1;
        const __bf16* hinB = (t & 1) ? hB1 : hB0;
        __bf16*      houtB = (t & 1) ? hB0 : hB1;
        k_lstm_step<<<dim3(DD / 32, 1, 2), 256, 0, stream>>>(
            whhF_bf, whhB_bf, xgF, xgB, hinF, hinB, houtF, houtB,
            cF, cB, hsF, hsB, t);
    }

    // --- classifier + CRF loss ---
    k_emissions<<<(MS * TT + 255) / 256, 256, 0, stream>>>(hsF, hsB, wcls, bcls, emis);
    k_crf<<<1, 32, 0, stream>>>(emis, tags, mask, trans, st, et, (float*)d_out);
}
